// InverseCognitiveProjector_3977139716532
// MI455X (gfx1250) — matrix operand order repaired and checked by reading, not compiled, not run on hardware
//
#include <hip/hip_runtime.h>
#include <math.h>

// ---------------- problem constants ----------------
#define B_ROWS 4096
#define E_DIM  64
#define H_DIM  128
#define V_DIM  50257
#define CHUNK  512
#define NCHUNK 99              // ceil(50257/512)
#define VPAD   (CHUNK*NCHUNK)  // 50688
#define TOPK   10

typedef __attribute__((ext_vector_type(8)))  float    v8f;
typedef __attribute__((ext_vector_type(16))) __bf16   v16bf;
typedef __attribute__((ext_vector_type(4)))  unsigned u32x4;
typedef __attribute__((ext_vector_type(8)))  int      i32x8;
typedef __attribute__((ext_vector_type(4)))  int      i32x4;

static __device__ inline __bf16 to_bf16(float f) {
    unsigned u = __builtin_bit_cast(unsigned, f);
    unsigned r = (u + 0x7FFFu + ((u >> 16) & 1u)) >> 16;
    unsigned short h = (unsigned short)r;
    return __builtin_bit_cast(__bf16, h);
}

// 16x32 bf16 A-fragment K index (ISA 7.12.2, 16-bit A layout).
static __device__ inline int a_frag_k(int j, int half) {
    int v = j >> 1, pos = j & 1;
    return (v < 4) ? (8 * half + 2 * v + pos) : (16 + 8 * half + 2 * (v - 4) + pos);
}

// ---------------- kernel 1: transpose + f32 -> bf16 ----------------
__global__ void transpose_to_bf16(const float* __restrict__ in, __bf16* __restrict__ out,
                                  int K, int N, long total) {
    long idx = (long)blockIdx.x * blockDim.x + threadIdx.x;
    if (idx >= total) return;
    int  k = (int)(idx % K);
    long n = idx / K;
    out[idx] = (n < N) ? to_bf16(in[(size_t)k * N + (size_t)n]) : to_bf16(0.0f);
}

// ---------------- kernel 2: fused 3-layer MLP (Linear+LN+ReLU) ----------------
__global__ __launch_bounds__(256) void mlp_kernel(
    const float* __restrict__ psi,
    const __bf16* __restrict__ W0T, const float* __restrict__ b0,
    const float* __restrict__ g0,  const float* __restrict__ be0,
    const __bf16* __restrict__ W1T, const float* __restrict__ b1,
    const float* __restrict__ g1,  const float* __restrict__ be1,
    const __bf16* __restrict__ W2T, const float* __restrict__ b2,
    const float* __restrict__ g2,  const float* __restrict__ be2,
    __bf16* __restrict__ x2_out) {
    __shared__ float  actF[16][H_DIM];
    __shared__ __bf16 actB[16][H_DIM];
    __shared__ float  mu_s[16], rs_s[16];

    const int tid  = threadIdx.x;
    const int lane = tid & 31, wv = tid >> 5;
    const int l15  = lane & 15, half = lane >> 4;
    const int row0 = blockIdx.x * 16;

    for (int i = tid; i < 16 * E_DIM; i += 256) {
        int r = i >> 6, c = i & 63;
        actB[r][c] = to_bf16(psi[(size_t)(row0 + r) * E_DIM + c]);
    }
    __syncthreads();

    for (int layer = 0; layer < 3; ++layer) {
        const int K = (layer == 0) ? E_DIM : H_DIM;
        const __bf16* WT = (layer == 0) ? W0T : ((layer == 1) ? W1T : W2T);
        const float* bb  = (layer == 0) ? b0  : ((layer == 1) ? b1  : b2);
        const float* gg  = (layer == 0) ? g0  : ((layer == 1) ? g1  : g2);
        const float* bee = (layer == 0) ? be0 : ((layer == 1) ? be1 : be2);

        const int n0 = wv * 16;
        v8f c = {};
        for (int s = 0; s < K / 32; ++s) {
            v16bf a;
#pragma unroll
            for (int j = 0; j < 16; ++j) a[j] = actB[l15][32 * s + 16 * half + j];
            v16bf b = *(const v16bf*)(WT + (size_t)(n0 + l15) * K + 32 * s + 16 * half);
            c = __builtin_amdgcn_wmma_f32_16x16x32_bf16(false, a, false, b,
                                                        (short)0, c, false, false);
        }
        __syncthreads();
        {
            float bias = bb[n0 + l15];
#pragma unroll
            for (int r = 0; r < 8; ++r) actF[r + 8 * half][n0 + l15] = c[r] + bias;
        }
        __syncthreads();
        if (tid < 16) {
            float m = 0.f;
            for (int j = 0; j < H_DIM; ++j) m += actF[tid][j];
            m *= (1.0f / H_DIM);
            float v = 0.f;
            for (int j = 0; j < H_DIM; ++j) { float d = actF[tid][j] - m; v += d * d; }
            v *= (1.0f / H_DIM);
            mu_s[tid] = m;
            rs_s[tid] = rsqrtf(v + 1e-5f);
        }
        __syncthreads();
        for (int i = tid; i < 16 * H_DIM; i += 256) {
            int r = i >> 7, cc = i & 127;
            float h = (actF[r][cc] - mu_s[r]) * rs_s[r] * gg[cc] + bee[cc];
            actB[r][cc] = to_bf16(h > 0.f ? h : 0.f);
        }
        __syncthreads();
    }
    for (int i = tid; i < 16 * H_DIM; i += 256) {
        int r = i >> 7, cc = i & 127;
        x2_out[(size_t)(row0 + r) * H_DIM + cc] = actB[r][cc];
    }
}

// ---------------- kernel 3: vocab GEMM + per-chunk top-10 ----------------
// grid = (NCHUNK, B/16); block 256 = 8 waves; each wave does 4 16x16 C tiles.
// A tile (16x128 bf16) staged into LDS by the Tensor Data Mover; B fragments
// double-buffered in registers to overlap L2 loads with WMMA issue.
__global__ __launch_bounds__(256) void logits_topk_kernel(
    const __bf16* __restrict__ x2, const __bf16* __restrict__ WvT,
    const float* __restrict__ bv,
    float* __restrict__ part_v, int* __restrict__ part_i) {
    __shared__ float  lg[16][CHUNK];        // 32 KB logits tile
    __shared__ __bf16 x2t[16][H_DIM];       // 4 KB activation tile (TDM dest)
    __shared__ float  tv[16][16][TOPK];     // 10 KB
    __shared__ int    ti[16][16][TOPK];     // 10 KB

    const int tid  = threadIdx.x;
    const int lane = tid & 31, wv = tid >> 5;
    const int l15  = lane & 15, half = lane >> 4;
    const int row0  = blockIdx.y * 16;
    const int nbase = blockIdx.x * CHUNK;

    __builtin_prefetch(WvT + (size_t)(nbase + wv * 64) * H_DIM, 0, 0);

    // ---- TDM: stage x2[row0:row0+16, 0:128] bf16 tile into LDS (wave 0) ----
    if (wv == 0) {
        unsigned long long ga =
            (unsigned long long)(size_t)(x2 + (size_t)row0 * H_DIM);
        unsigned lds_addr =
            (unsigned)(size_t)(&x2t[0][0]);   // low 32 bits = LDS byte offset
        u32x4 g0;
        g0[0] = 1u;                                          // count=1
        g0[1] = lds_addr;                                    // lds_addr
        g0[2] = (unsigned)(ga & 0xFFFFFFFFu);                // global_addr lo
        g0[3] = (unsigned)((ga >> 32) & 0x1FFFFFFu)          // global_addr hi
                | (2u << 30);                                // type=2 (image)
        i32x8 g1;
        g1[0] = (1 << 16);                 // workgroup_mask=0, data_size=1 (2B)
        g1[1] = (int)(H_DIM << 16);        // tensor_dim0 lo16 (=128)
        g1[2] = (int)((unsigned)(B_ROWS & 0xFFFF) << 16);  // dim0 hi=0 | dim1 lo16
        g1[3] = (int)(((B_ROWS >> 16) & 0xFFFF) | (H_DIM << 16)); // dim1 hi | tile_dim0=128
        g1[4] = 16;                        // tile_dim1=16, tile_dim2=0
        g1[5] = H_DIM;                     // tensor_dim0_stride lo32 = 128
        g1[6] = 0;                         // stride0 hi16 | stride1 lo16
        g1[7] = 0;                         // stride1 hi32
        i32x4 z4 = {0, 0, 0, 0};
        i32x8 z8 = {0, 0, 0, 0, 0, 0, 0, 0};
        __builtin_amdgcn_tensor_load_to_lds(g0, g1, z4, z4, z8, 0);
        __builtin_amdgcn_s_wait_tensorcnt(0);
    }
    __syncthreads();

    // A fragments for K=128 (4 bf16 WMMA K-steps) from the LDS tile
    v16bf a[4];
#pragma unroll
    for (int s = 0; s < 4; ++s)
#pragma unroll
        for (int j = 0; j < 16; ++j) a[s][j] = x2t[l15][32 * s + 16 * half + j];

    // Double-buffered B fragments across the 4 column subtiles
    v16bf bb[2][4];
    {
        const __bf16* bp = WvT + (size_t)(nbase + wv * 64 + l15) * H_DIM + 16 * half;
#pragma unroll
        for (int s = 0; s < 4; ++s) bb[0][s] = *(const v16bf*)(bp + 32 * s);
    }
#pragma unroll
    for (int t = 0; t < 4; ++t) {
        if (t < 3) {  // issue next tile's loads before consuming current tile
            const __bf16* bp =
                WvT + (size_t)(nbase + wv * 64 + (t + 1) * 16 + l15) * H_DIM + 16 * half;
#pragma unroll
            for (int s = 0; s < 4; ++s) bb[(t + 1) & 1][s] = *(const v16bf*)(bp + 32 * s);
        }
        v8f c = {};
#pragma unroll
        for (int s = 0; s < 4; ++s)
            c = __builtin_amdgcn_wmma_f32_16x16x32_bf16(false, a[s], false, bb[t & 1][s],
                                                        (short)0, c, false, false);
        const int n0 = nbase + wv * 64 + t * 16;
        const int n  = n0 + l15;
        const float bias = (n < V_DIM) ? bv[n] : -3.0e38f;  // padded cols never win
#pragma unroll
        for (int r = 0; r < 8; ++r) lg[r + 8 * half][n0 - nbase + l15] = c[r] + bias;
    }
    __syncthreads();

    // 16 threads per row, each scans 32 columns keeping a sorted-desc top-10
    {
        const int r = tid >> 4, sub = tid & 15;
        float bv10[TOPK]; int bi10[TOPK];
#pragma unroll
        for (int k = 0; k < TOPK; ++k) { bv10[k] = -3.4e38f; bi10[k] = 0; }
        for (int j = 0; j < 32; ++j) {
            const int col = sub * 32 + j;
            const float val = lg[r][col];
            if (val > bv10[TOPK - 1]) {
                int p = TOPK - 1;
                while (p > 0 && bv10[p - 1] < val) {
                    bv10[p] = bv10[p - 1]; bi10[p] = bi10[p - 1]; --p;
                }
                bv10[p] = val; bi10[p] = nbase + col;
            }
        }
#pragma unroll
        for (int k = 0; k < TOPK; ++k) { tv[r][sub][k] = bv10[k]; ti[r][sub][k] = bi10[k]; }
    }
    __syncthreads();

    if (tid < 16) {  // merge 16 sorted lists -> chunk top-10
        float bv10[TOPK]; int bi10[TOPK];
#pragma unroll
        for (int k = 0; k < TOPK; ++k) { bv10[k] = -3.4e38f; bi10[k] = 0; }
        for (int sub = 0; sub < 16; ++sub)
            for (int k = 0; k < TOPK; ++k) {
                const float val = tv[tid][sub][k];
                if (val <= bv10[TOPK - 1]) break;  // sorted desc per sub-list
                int p = TOPK - 1;
                while (p > 0 && bv10[p - 1] < val) {
                    bv10[p] = bv10[p - 1]; bi10[p] = bi10[p - 1]; --p;
                }
                bv10[p] = val; bi10[p] = ti[tid][sub][k];
            }
        const size_t base = ((size_t)(row0 + tid) * NCHUNK + blockIdx.x) * TOPK;
#pragma unroll
        for (int k = 0; k < TOPK; ++k) { part_v[base + k] = bv10[k]; part_i[base + k] = bi10[k]; }
    }
}

// ---------------- kernel 4: final merge + softmax weights + gather-combine ----
__global__ __launch_bounds__(256) void combine_kernel(
    const float* __restrict__ part_v, const int* __restrict__ part_i,
    const float* __restrict__ qv, float* __restrict__ out) {
    __shared__ float wn[TOPK];
    __shared__ int   wi[TOPK];
    const int row = blockIdx.x, tid = threadIdx.x;

    if (tid == 0) {
        float bv10[TOPK]; int bi10[TOPK];
#pragma unroll
        for (int k = 0; k < TOPK; ++k) { bv10[k] = -3.4e38f; bi10[k] = 0; }
        for (int c = 0; c < NCHUNK; ++c) {
            const size_t base = ((size_t)row * NCHUNK + c) * TOPK;
            for (int k = 0; k < TOPK; ++k) {
                const float val = part_v[base + k];
                if (val <= bv10[TOPK - 1]) break;  // lists sorted desc
                int p = TOPK - 1;
                while (p > 0 && bv10[p - 1] < val) {
                    bv10[p] = bv10[p - 1]; bi10[p] = bi10[p - 1]; --p;
                }
                bv10[p] = val; bi10[p] = part_i[base + k];
            }
        }
        // softmax denominator cancels: weights renormalized over the kept 10
        const float m = bv10[0];
        float w[TOPK], tot = 0.f;
#pragma unroll
        for (int k = 0; k < TOPK; ++k) { w[k] = __expf(bv10[k] - m); tot += w[k]; }
        const float inv = 1.0f / tot;
#pragma unroll
        for (int k = 0; k < TOPK; ++k) { wn[k] = w[k] * inv; wi[k] = bi10[k]; }
    }
    __syncthreads();

    float acc = 0.f;
#pragma unroll
    for (int k = 0; k < TOPK; ++k)
        acc += wn[k] * qv[(size_t)wi[k] * (E_DIM * 4) + tid];
    out[(size_t)row * (E_DIM * 4) + tid] = acc;
}

// ---------------- host launch ----------------
extern "C" void kernel_launch(void* const* d_in, const int* in_sizes, int n_in,
                              void* d_out, int out_size, void* d_ws, size_t ws_size,
                              hipStream_t stream) {
    (void)in_sizes; (void)n_in; (void)out_size; (void)ws_size;
    const float* psi = (const float*)d_in[0];
    const float* qv  = (const float*)d_in[1];
    const float* W0  = (const float*)d_in[2];
    const float* b0  = (const float*)d_in[3];
    const float* g0  = (const float*)d_in[4];
    const float* be0 = (const float*)d_in[5];
    const float* W1  = (const float*)d_in[6];
    const float* b1  = (const float*)d_in[7];
    const float* g1  = (const float*)d_in[8];
    const float* be1 = (const float*)d_in[9];
    const float* W2  = (const float*)d_in[10];
    const float* b2  = (const float*)d_in[11];
    const float* g2  = (const float*)d_in[12];
    const float* be2 = (const float*)d_in[13];
    const float* Wv  = (const float*)d_in[14];
    const float* bv  = (const float*)d_in[15];
    float* out = (float*)d_out;

    char* ws = (char*)d_ws;
    size_t off = 0;
    __bf16* WvT = (__bf16*)(ws + off); off += (size_t)VPAD * H_DIM * 2;     // 12.98 MB
    __bf16* W0T = (__bf16*)(ws + off); off += (size_t)H_DIM * E_DIM * 2;    // 16 KB
    __bf16* W1T = (__bf16*)(ws + off); off += (size_t)H_DIM * H_DIM * 2;    // 32 KB
    __bf16* W2T = (__bf16*)(ws + off); off += (size_t)H_DIM * H_DIM * 2;    // 32 KB
    __bf16* x2b = (__bf16*)(ws + off); off += (size_t)B_ROWS * H_DIM * 2;   // 1 MB
    float*  pv  = (float*)(ws + off);  off += (size_t)B_ROWS * NCHUNK * TOPK * 4; // 16.2 MB
    int*    pi  = (int*)(ws + off);    off += (size_t)B_ROWS * NCHUNK * TOPK * 4; // 16.2 MB

    {
        long tot = (long)H_DIM * E_DIM;  // W0T: [128][64]
        transpose_to_bf16<<<(unsigned)((tot + 255) / 256), 256, 0, stream>>>(W0, W0T, E_DIM, H_DIM, tot);
        tot = (long)H_DIM * H_DIM;
        transpose_to_bf16<<<(unsigned)((tot + 255) / 256), 256, 0, stream>>>(W1, W1T, H_DIM, H_DIM, tot);
        transpose_to_bf16<<<(unsigned)((tot + 255) / 256), 256, 0, stream>>>(W2, W2T, H_DIM, H_DIM, tot);
        tot = (long)VPAD * H_DIM;        // WvT: [50688][128], pad rows zero
        transpose_to_bf16<<<(unsigned)((tot + 255) / 256), 256, 0, stream>>>(Wv, WvT, H_DIM, V_DIM, tot);
    }

    mlp_kernel<<<B_ROWS / 16, 256, 0, stream>>>(psi, W0T, b0, g0, be0,
                                                W1T, b1, g1, be1,
                                                W2T, b2, g2, be2, x2b);

    logits_topk_kernel<<<dim3(NCHUNK, B_ROWS / 16), 256, 0, stream>>>(x2b, WvT, bv, pv, pi);

    combine_kernel<<<B_ROWS, 256, 0, stream>>>(pv, pi, qv, out);
}
